// GraphTransformer_90666759619162
// MI455X (gfx1250) — matrix operand order repaired and checked by reading, not compiled, not run on hardware
//
#include <hip/hip_runtime.h>

// ---------------- problem constants (from the reference) ----------------
#define NN   50000
#define EE   800000
#define IN_F 128
#define NH   8
#define HIDD 32
#define HD   256      // NH * HIDD
#define OUTF 64
#define LN_EPS 1e-5f
#define INV_SQRT_HID 0.17677669529663687f   // 1/sqrt(32)

typedef __attribute__((ext_vector_type(16))) __bf16       v16bf;
typedef __attribute__((ext_vector_type(8)))  float        v8f;
typedef __attribute__((ext_vector_type(4)))  unsigned int u32x4;
typedef __attribute__((ext_vector_type(4)))  float        f32x4;

// ---------------- helpers ----------------
__device__ __forceinline__ __bf16 tobf(float f) {
    unsigned u = __float_as_uint(f);
    u += 0x7fffu + ((u >> 16) & 1u);            // round-to-nearest-even
    unsigned short h = (unsigned short)(u >> 16);
    __bf16 r; __builtin_memcpy(&r, &h, 2); return r;
}
// monotone float<->signed-int encoding for atomicMax on floats
__device__ __forceinline__ int enc_f(float f) {
    int i = __float_as_int(f); return i >= 0 ? i : (i ^ 0x7fffffff);
}
__device__ __forceinline__ float dec_f(int i) {
    return __int_as_float(i >= 0 ? i : (i ^ 0x7fffffff));
}
__device__ __forceinline__ void atomAddF(float* p, float v) {
    __hip_atomic_fetch_add(p, v, __ATOMIC_RELAXED, __HIP_MEMORY_SCOPE_AGENT);
}

// ---------------- conversion kernels ----------------
__global__ void k_conv_act(const float* __restrict__ in, __bf16* __restrict__ out, int n) {
    int i = blockIdx.x * blockDim.x + threadIdx.x;
    if (i < n) out[i] = tobf(in[i]);
}
// W[K][Nout] fp32 -> WT[Nout][K] bf16 (so B fragments are contiguous-in-K)
__global__ void k_conv_wT(const float* __restrict__ W, __bf16* __restrict__ WT, int K, int Nout) {
    int i = blockIdx.x * blockDim.x + threadIdx.x;
    if (i >= K * Nout) return;
    int k = i / Nout, n = i - k * Nout;
    WT[(size_t)n * K + k] = tobf(W[i]);
}

// ---------------- WMMA GEMM: C[M][Nout] = A[M][K](bf16) @ WT[Nout][K]^T + bias ----------------
// one wave per 16x16 tile; K-loop in steps of 32 using v_wmma_f32_16x16x32_bf16
__global__ void k_wmma_gemm(const __bf16* __restrict__ A, const __bf16* __restrict__ BT,
                            const float* __restrict__ bias, float* __restrict__ C,
                            int K, int Nout, int ntiles, int total_tiles) {
    int wave = blockIdx.x * (blockDim.x >> 5) + (threadIdx.x >> 5);
    if (wave >= total_tiles) return;                 // wave-uniform: EXEC all-ones inside
    int lane  = threadIdx.x & 31;
    int mtile = wave / ntiles;
    int ntile = wave - mtile * ntiles;
    int l15   = lane & 15;
    int g     = lane >> 4;                            // half-wave select per ISA layout
    int ncol  = ntile * 16 + l15;
    const __bf16* arow = A  + (size_t)(mtile * 16 + l15) * K;
    const __bf16* brow = BT + (size_t)ncol * K;

    v8f c = {};
    for (int k0 = 0; k0 < K; k0 += 32) {
        union { v16bf v; u32x4 u[2]; } a, b;
        // A 16x32 bf16 fragment: lane holds K = {8g..8g+7, 16+8g..16+8g+7} of its row
        a.u[0] = *(const u32x4*)(arow + k0 + 8 * g);
        a.u[1] = *(const u32x4*)(arow + k0 + 16 + 8 * g);
        // B 32x16 fragment: lane = column ncol, K = {8g..8g+7, 16+8g..16+8g+7} (contiguous in WT)
        b.u[0] = *(const u32x4*)(brow + k0 + 8 * g);
        b.u[1] = *(const u32x4*)(brow + k0 + 16 + 8 * g);
        c = __builtin_amdgcn_wmma_f32_16x16x32_bf16(false, a.v, false, b.v,
                                                    (short)0, c, false, false);
    }
    float bv = bias ? bias[ncol] : 0.0f;
    int mbase = mtile * 16 + 8 * g;                   // C/D layout: VGPR r -> M = 8g + r
    #pragma unroll
    for (int r = 0; r < 8; ++r)
        C[(size_t)(mbase + r) * Nout + ncol] = c[r] + bv;
}

// ---------------- edge kernels (wave per edge) ----------------
__global__ void k_init_ms(int* __restrict__ m, float* __restrict__ s, int n) {
    int i = blockIdx.x * blockDim.x + threadIdx.x;
    if (i < n) { m[i] = 0x80000000; s[i] = 0.0f; }
}

// score = dot(q[dst,h,:], k[src,h,:] + attr*We[h,:]) / sqrt(32); atomic segment-max
__global__ void k_edge_score(const float* __restrict__ q, const float* __restrict__ k,
                             const long long* __restrict__ ei, const float* __restrict__ eattr,
                             const float* __restrict__ We, float* __restrict__ scores,
                             int* __restrict__ menc, int E) {
    int e = blockIdx.x * (blockDim.x >> 5) + (threadIdx.x >> 5);
    if (e >= E) return;
    int lane = threadIdx.x & 31;
    long long src = ei[e], dst = ei[(size_t)E + e];
    int h = lane >> 2, part = lane & 3;
    int base = h * HIDD + part * 8;
    float attr = eattr[e];
    f32x4 q0 = *(const f32x4*)(q + (size_t)dst * HD + base);
    f32x4 q1 = *(const f32x4*)(q + (size_t)dst * HD + base + 4);
    f32x4 k0 = *(const f32x4*)(k + (size_t)src * HD + base);
    f32x4 k1 = *(const f32x4*)(k + (size_t)src * HD + base + 4);
    f32x4 w0 = *(const f32x4*)(We + base);
    f32x4 w1 = *(const f32x4*)(We + base + 4);
    float d;
    d  = q0.x * fmaf(attr, w0.x, k0.x);
    d += q0.y * fmaf(attr, w0.y, k0.y);
    d += q0.z * fmaf(attr, w0.z, k0.z);
    d += q0.w * fmaf(attr, w0.w, k0.w);
    d += q1.x * fmaf(attr, w1.x, k1.x);
    d += q1.y * fmaf(attr, w1.y, k1.y);
    d += q1.z * fmaf(attr, w1.z, k1.z);
    d += q1.w * fmaf(attr, w1.w, k1.w);
    d += __shfl_xor(d, 1, 32);                 // reduce the 4 lanes of this head
    d += __shfl_xor(d, 2, 32);
    if (part == 0) {
        float sc = d * INV_SQRT_HID;
        scores[(size_t)e * NH + h] = sc;
        atomicMax(&menc[(size_t)dst * NH + h], enc_f(sc));
    }
}

// ex = exp(score - m[dst]); scores <- ex; s[dst] += ex   (thread per (edge,head))
__global__ void k_edge_expsum(const long long* __restrict__ ei, float* __restrict__ scores,
                              const int* __restrict__ menc, float* __restrict__ ssum, int E) {
    int idx = blockIdx.x * blockDim.x + threadIdx.x;
    if (idx >= E * NH) return;
    int e = idx >> 3, h = idx & 7;
    long long dst = ei[(size_t)E + e];
    float mm = dec_f(menc[(size_t)dst * NH + h]);
    float ex = __expf(scores[idx] - mm);
    scores[idx] = ex;
    atomAddF(&ssum[(size_t)dst * NH + h], ex);
}

// agg[dst] += alpha * (v[src] + attr*We)   (wave per edge; agg pre-seeded with skip term)
__global__ void k_edge_message(const float* __restrict__ v, const long long* __restrict__ ei,
                               const float* __restrict__ eattr, const float* __restrict__ We,
                               const float* __restrict__ scores, const float* __restrict__ ssum,
                               float* __restrict__ agg, int E) {
    int e = blockIdx.x * (blockDim.x >> 5) + (threadIdx.x >> 5);
    if (e >= E) return;
    int lane = threadIdx.x & 31;
    long long src = ei[e], dst = ei[(size_t)E + e];
    int h = lane >> 2, part = lane & 3;
    int base = h * HIDD + part * 8;
    float attr  = eattr[e];
    float alpha = scores[(size_t)e * NH + h] / ssum[(size_t)dst * NH + h];
    f32x4 v0 = *(const f32x4*)(v + (size_t)src * HD + base);
    f32x4 v1 = *(const f32x4*)(v + (size_t)src * HD + base + 4);
    f32x4 w0 = *(const f32x4*)(We + base);
    f32x4 w1 = *(const f32x4*)(We + base + 4);
    float* o = agg + (size_t)dst * HD + base;
    atomAddF(o + 0, alpha * fmaf(attr, w0.x, v0.x));
    atomAddF(o + 1, alpha * fmaf(attr, w0.y, v0.y));
    atomAddF(o + 2, alpha * fmaf(attr, w0.z, v0.z));
    atomAddF(o + 3, alpha * fmaf(attr, w0.w, v0.w));
    atomAddF(o + 4, alpha * fmaf(attr, w1.x, v1.x));
    atomAddF(o + 5, alpha * fmaf(attr, w1.y, v1.y));
    atomAddF(o + 6, alpha * fmaf(attr, w1.z, v1.z));
    atomAddF(o + 7, alpha * fmaf(attr, w1.w, v1.w));
}

// ---------------- graph layernorm (global mean/std) + ELU -> bf16 ----------------
__global__ void k_zero2(float* s) { if (threadIdx.x < 2) s[threadIdx.x] = 0.0f; }

__global__ void k_ln_reduce(const float* __restrict__ x, float* __restrict__ stats, int n) {
    __shared__ float ssum[256], ssq[256];
    float s = 0.0f, q = 0.0f;
    for (int i = blockIdx.x * blockDim.x + threadIdx.x; i < n; i += gridDim.x * blockDim.x) {
        float vv = x[i]; s += vv; q = fmaf(vv, vv, q);
    }
    ssum[threadIdx.x] = s; ssq[threadIdx.x] = q;
    __syncthreads();
    for (int off = 128; off > 0; off >>= 1) {
        if (threadIdx.x < off) {
            ssum[threadIdx.x] += ssum[threadIdx.x + off];
            ssq[threadIdx.x]  += ssq[threadIdx.x + off];
        }
        __syncthreads();
    }
    if (threadIdx.x == 0) { atomAddF(&stats[0], ssum[0]); atomAddF(&stats[1], ssq[0]); }
}

__global__ void k_ln_apply_elu(const float* __restrict__ x, const float* __restrict__ stats,
                               const float* __restrict__ w, const float* __restrict__ b,
                               __bf16* __restrict__ out, int n, float inv_count) {
    int i = blockIdx.x * blockDim.x + threadIdx.x;
    if (i >= n) return;
    float mean = stats[0] * inv_count;
    float var  = fmaf(-mean, mean, stats[1] * inv_count);
    float inv  = 1.0f / (sqrtf(fmaxf(var, 0.0f)) + LN_EPS);
    int col = i & (HD - 1);
    float y = fmaf((x[i] - mean) * inv, w[col], b[col]);
    y = y > 0.0f ? y : (__expf(y) - 1.0f);              // ELU
    out[i] = tobf(y);
}

// ---------------- host orchestration ----------------
extern "C" void kernel_launch(void* const* d_in, const int* in_sizes, int n_in,
                              void* d_out, int out_size, void* d_ws, size_t ws_size,
                              hipStream_t stream) {
    (void)in_sizes; (void)n_in; (void)out_size; (void)ws_size;
    const float*     x     = (const float*)d_in[0];
    const long long* ei    = (const long long*)d_in[1];
    const float*     eattr = (const float*)d_in[2];
    const float *Wq1=(const float*)d_in[3],  *bq1=(const float*)d_in[4];
    const float *Wk1=(const float*)d_in[5],  *bk1=(const float*)d_in[6];
    const float *Wv1=(const float*)d_in[7],  *bv1=(const float*)d_in[8];
    const float *We1=(const float*)d_in[9];
    const float *Ws1=(const float*)d_in[10], *bs1=(const float*)d_in[11];
    const float *lnw1=(const float*)d_in[12],*lnb1=(const float*)d_in[13];
    const float *Wq2=(const float*)d_in[14], *bq2=(const float*)d_in[15];
    const float *Wk2=(const float*)d_in[16], *bk2=(const float*)d_in[17];
    const float *Wv2=(const float*)d_in[18], *bv2=(const float*)d_in[19];
    const float *We2=(const float*)d_in[20];
    const float *Ws2=(const float*)d_in[21], *bs2=(const float*)d_in[22];
    const float *lnw2=(const float*)d_in[23],*lnb2=(const float*)d_in[24];
    const float *Wo =(const float*)d_in[25], *bo =(const float*)d_in[26];

    // ---- workspace carve (256B aligned) ----
    char* p = (char*)d_ws;
    auto carve = [&](size_t bytes) { char* r = p; p += (bytes + 255) & ~(size_t)255; return r; };
    __bf16* xb   = (__bf16*)carve((size_t)NN * IN_F * 2);
    __bf16* hb   = (__bf16*)carve((size_t)NN * HD * 2);
    float*  qb   = (float*) carve((size_t)NN * HD * 4);
    float*  kb   = (float*) carve((size_t)NN * HD * 4);
    float*  vb   = (float*) carve((size_t)NN * HD * 4);
    float*  agg  = (float*) carve((size_t)NN * HD * 4);
    float*  sc   = (float*) carve((size_t)EE * NH * 4);
    int*    menc = (int*)   carve((size_t)NN * NH * 4);
    float*  ssum = (float*) carve((size_t)NN * NH * 4);
    float*  stats= (float*) carve(64);
    __bf16* WqT1=(__bf16*)carve((size_t)IN_F*HD*2), *WkT1=(__bf16*)carve((size_t)IN_F*HD*2);
    __bf16* WvT1=(__bf16*)carve((size_t)IN_F*HD*2), *WsT1=(__bf16*)carve((size_t)IN_F*HD*2);
    __bf16* WqT2=(__bf16*)carve((size_t)HD*HD*2),   *WkT2=(__bf16*)carve((size_t)HD*HD*2);
    __bf16* WvT2=(__bf16*)carve((size_t)HD*HD*2),   *WsT2=(__bf16*)carve((size_t)HD*HD*2);
    __bf16* WoT =(__bf16*)carve((size_t)HD*OUTF*2);

    auto convW = [&](const float* W, __bf16* WT, int K, int Nout) {
        int n = K * Nout;
        k_conv_wT<<<(n + 255) / 256, 256, 0, stream>>>(W, WT, K, Nout);
    };
    auto gemm = [&](const __bf16* A, const __bf16* BT, const float* bias, float* C,
                    int K, int Nout) {
        int nt = Nout / 16, tot = (NN / 16) * nt;
        k_wmma_gemm<<<(tot + 7) / 8, 256, 0, stream>>>(A, BT, bias, C, K, Nout, nt, tot);
    };
    auto conv_edges = [&](const float* We) {
        k_init_ms<<<(NN * NH + 255) / 256, 256, 0, stream>>>(menc, ssum, NN * NH);
        k_edge_score<<<(EE + 7) / 8, 256, 0, stream>>>(qb, kb, ei, eattr, We, sc, menc, EE);
        k_edge_expsum<<<(EE * NH + 255) / 256, 256, 0, stream>>>(ei, sc, menc, ssum, EE);
        k_edge_message<<<(EE + 7) / 8, 256, 0, stream>>>(vb, ei, eattr, We, sc, ssum, agg, EE);
    };
    auto ln_elu = [&](const float* w, const float* b) {
        k_zero2<<<1, 32, 0, stream>>>(stats);
        k_ln_reduce<<<1024, 256, 0, stream>>>(agg, stats, NN * HD);
        k_ln_apply_elu<<<(NN * HD + 255) / 256, 256, 0, stream>>>(
            agg, stats, w, b, hb, NN * HD, 1.0f / ((float)NN * HD));
    };

    // ---- weight / activation conversion (bf16, weights transposed) ----
    k_conv_act<<<(NN * IN_F + 255) / 256, 256, 0, stream>>>(x, xb, NN * IN_F);
    convW(Wq1, WqT1, IN_F, HD); convW(Wk1, WkT1, IN_F, HD);
    convW(Wv1, WvT1, IN_F, HD); convW(Ws1, WsT1, IN_F, HD);
    convW(Wq2, WqT2, HD, HD);   convW(Wk2, WkT2, HD, HD);
    convW(Wv2, WvT2, HD, HD);   convW(Ws2, WsT2, HD, HD);
    convW(Wo,  WoT,  HD, OUTF);

    // ---- layer 1 ----
    gemm(xb, WqT1, bq1, qb, IN_F, HD);
    gemm(xb, WkT1, bk1, kb, IN_F, HD);
    gemm(xb, WvT1, bv1, vb, IN_F, HD);
    gemm(xb, WsT1, bs1, agg, IN_F, HD);     // skip term seeds the aggregation buffer
    conv_edges(We1);
    ln_elu(lnw1, lnb1);                      // agg -> hb (bf16)

    // ---- layer 2 ----
    gemm(hb, WqT2, bq2, qb, HD, HD);
    gemm(hb, WkT2, bk2, kb, HD, HD);
    gemm(hb, WvT2, bv2, vb, HD, HD);
    gemm(hb, WsT2, bs2, agg, HD, HD);
    conv_edges(We2);
    ln_elu(lnw2, lnb2);                      // agg -> hb (bf16)

    // ---- output projection ----
    gemm(hb, WoT, bo, (float*)d_out, HD, OUTF);
}
